// ConvAE_16801912062302
// MI455X (gfx1250) — matrix operand order repaired and checked by reading, not compiled, not run on hardware
//
#include <hip/hip_runtime.h>

typedef __attribute__((ext_vector_type(16))) __bf16 v16bf;
typedef __attribute__((ext_vector_type(8)))  __bf16 v8bf;
typedef __attribute__((ext_vector_type(4)))  __bf16 v4bf;
typedef __attribute__((ext_vector_type(8)))  float  v8f;

constexpr int kB = 16;      // batch
constexpr int kP = 4096;    // positions
constexpr int kD = 256;     // feature dim
constexpr int kC = 512;     // channels

constexpr int NC   = 32;          // channels per workgroup (2 waves x 16)
constexpr int CB   = kC / NC;     // 16 channel blocks
constexpr int PCH  = 256;         // positions per chunk
constexpr int NPCH = kP / PCH;    // 16 chunks
constexpr int NSTR = PCH / 16;    // 16 strips of 16 positions
constexpr int KCH  = kD / 32;     // 8 K-chunks of 32 for bf16 WMMA

// ---------------------------------------------------------------------------
// Kernel 0: one-shot split of x into planar bf16 hi/lo (written into d_out,
// which is consumed by the GEMM before being zero-filled for the real output).
// ---------------------------------------------------------------------------
__global__ void split_x_kernel(const float4* __restrict__ x4,
                               __bf16* __restrict__ xh,
                               __bf16* __restrict__ xl, int n4)
{
    const int i = blockIdx.x * blockDim.x + threadIdx.x;
    if (i >= n4) return;
    const float4 v = x4[i];
    const float f[4] = {v.x, v.y, v.z, v.w};
    v4bf h, l;
    #pragma unroll
    for (int j = 0; j < 4; ++j) {
        const __bf16 hj = (__bf16)f[j];
        h[j] = hj;
        l[j] = (__bf16)(f[j] - (float)hj);
    }
    *(v4bf*)&xh[(size_t)i * 4] = h;   // 8-byte stores
    *(v4bf*)&xl[(size_t)i * 4] = l;
}

// ---------------------------------------------------------------------------
// Kernel 1: fused GEMM (bf16 split-precision WMMA) + per-chunk argmax.
// grid.x = kB*CB, grid.y = NPCH; block = 64 threads = 2 waves.
// Strips staged with GLOBAL_LOAD_ASYNC_TO_LDS_B128 (ASYNCcnt path);
// A fragments software-pipelined one K-chunk ahead; two accumulators.
// ---------------------------------------------------------------------------
__global__ __launch_bounds__(64) void gemm_argmax_kernel(
    const __bf16* __restrict__ xh,     // (kB, kP, kD) hi plane
    const __bf16* __restrict__ xl,     // (kB, kP, kD) lo plane
    const float* __restrict__ kern,    // (kC, kD)
    float* __restrict__ pmax,          // (kB*kC, NPCH)
    int*   __restrict__ pidx)          // (kB*kC, NPCH)
{
    __shared__ __bf16 skh[NC][kD];                 // kernels hi (16 KB)
    __shared__ __bf16 skl[NC][kD];                 // kernels lo (16 KB)
    __shared__ __align__(16) __bf16 sxh[16][kD];   // x strip hi (8 KB)
    __shared__ __align__(16) __bf16 sxl[16][kD];   // x strip lo (8 KB)

    const int b     = blockIdx.x / CB;
    const int c0    = (blockIdx.x % CB) * NC;
    const int pc    = blockIdx.y;
    const int pbase = pc * PCH;

    const int tid  = threadIdx.x;
    const int lane = tid & 31;
    const int wave = tid >> 5;
    const int half = lane >> 4;
    const int l16  = lane & 15;

    // one-time: load + hi/lo split the 32-channel kernel block into LDS
    for (int i = tid * 4; i < NC * kD; i += 64 * 4) {
        const float4 v = *(const float4*)&kern[c0 * kD + i];
        const float f[4] = {v.x, v.y, v.z, v.w};
        #pragma unroll
        for (int j = 0; j < 4; ++j) {
            const __bf16 h = (__bf16)f[j];
            (&skh[0][0])[i + j] = h;
            (&skl[0][0])[i + j] = (__bf16)(f[j] - (float)h);
        }
    }
    __syncthreads();

    const int cloc = wave * 16 + l16;   // this lane's channel within block

    // hoist B fragments for the whole chunk: the k order used here (k = 16*half + s, same as A) puts a
    // lane's 16 K-values contiguous -> one aligned 32B LDS read per chunk.
    v16bf bh[KCH], bl[KCH];
    #pragma unroll
    for (int k = 0; k < KCH; ++k) {
        bh[k] = *(const v16bf*)&skh[cloc][k * 32 + half * 16];
        bl[k] = *(const v16bf*)&skl[cloc][k * 32 + half * 16];
    }

    // workgroup-relative LDS byte offsets (low 32 bits of generic address)
    const unsigned sxh_off = (unsigned)(uintptr_t)&sxh[0][0];
    const unsigned sxl_off = (unsigned)(uintptr_t)&sxl[0][0];

    float bestV = -__builtin_inff();
    int   bestI = 0;

    for (int s = 0; s < NSTR; ++s) {
        const int p0 = pbase + s * 16;

        __syncthreads();   // previous strip's LDS reads complete

        // async direct-to-LDS staging: 512 x 16B per plane, 8 per thread
        {
            const __bf16* gh = &xh[((size_t)b * kP + p0) * kD];
            const __bf16* gl = &xl[((size_t)b * kP + p0) * kD];
            #pragma unroll
            for (int i = tid; i < (16 * kD * 2) / 16; i += 64) {
                asm volatile("global_load_async_to_lds_b128 %0, %1, off"
                             :: "v"(sxh_off + i * 16), "v"(gh + i * 8)
                             : "memory");
                asm volatile("global_load_async_to_lds_b128 %0, %1, off"
                             :: "v"(sxl_off + i * 16), "v"(gl + i * 8)
                             : "memory");
            }
            asm volatile("s_wait_asynccnt 0" ::: "memory");
        }
        __syncthreads();

        const __bf16* rowh = &sxh[l16][0];
        const __bf16* rowl = &sxl[l16][0];

        // software-pipelined A-fragment reads (one K-chunk ahead)
        v8bf ch0, ch1, cl0, cl1, nh0, nh1, nl0, nl1;
        {
            const int base = half * 16;
            ch0 = *(const v8bf*)(rowh + base);
            ch1 = *(const v8bf*)(rowh + base + 8);
            cl0 = *(const v8bf*)(rowl + base);
            cl1 = *(const v8bf*)(rowl + base + 8);
        }

        v8f acc0 = {}, acc1 = {};
        #pragma unroll
        for (int k = 0; k < KCH; ++k) {
            if (k + 1 < KCH) {
                const int nb = (k + 1) * 32 + half * 16;
                nh0 = *(const v8bf*)(rowh + nb);
                nh1 = *(const v8bf*)(rowh + nb + 8);
                nl0 = *(const v8bf*)(rowl + nb);
                nl1 = *(const v8bf*)(rowl + nb + 8);
            }
            const v16bf ah = __builtin_shufflevector(
                ch0, ch1, 0, 1, 2, 3, 4, 5, 6, 7, 8, 9, 10, 11, 12, 13, 14, 15);
            const v16bf al = __builtin_shufflevector(
                cl0, cl1, 0, 1, 2, 3, 4, 5, 6, 7, 8, 9, 10, 11, 12, 13, 14, 15);
            // split precision: hi*hi + hi*lo + lo*hi, exact f32 accumulate;
            // two accumulators halve the serial D-dependency chain.
            if ((k & 1) == 0) {
                acc0 = __builtin_amdgcn_wmma_f32_16x16x32_bf16(
                    false, ah, false, bh[k], (short)0, acc0, false, false);
                acc0 = __builtin_amdgcn_wmma_f32_16x16x32_bf16(
                    false, ah, false, bl[k], (short)0, acc0, false, false);
                acc0 = __builtin_amdgcn_wmma_f32_16x16x32_bf16(
                    false, al, false, bh[k], (short)0, acc0, false, false);
            } else {
                acc1 = __builtin_amdgcn_wmma_f32_16x16x32_bf16(
                    false, ah, false, bh[k], (short)0, acc1, false, false);
                acc1 = __builtin_amdgcn_wmma_f32_16x16x32_bf16(
                    false, ah, false, bl[k], (short)0, acc1, false, false);
                acc1 = __builtin_amdgcn_wmma_f32_16x16x32_bf16(
                    false, al, false, bh[k], (short)0, acc1, false, false);
            }
            ch0 = nh0; ch1 = nh1; cl0 = nl0; cl1 = nl1;
        }

        // C/D layout: lane n = column, VGPR r -> row M=r (+8 for lanes 16..31)
        #pragma unroll
        for (int r = 0; r < 8; ++r) {
            const int p = p0 + (half ? r + 8 : r);   // ascending per lane
            const float v = acc0[r] + acc1[r];
            if (v > bestV) { bestV = v; bestI = p; }
        }
    }

    // merge the two M-halves (lane n <-> lane n+16); tie -> smaller index
    const float oV = __shfl_xor(bestV, 16);
    const int   oI = __shfl_xor(bestI, 16);
    if (oV > bestV || (oV == bestV && oI < bestI)) { bestV = oV; bestI = oI; }

    if (half == 0) {
        const size_t off = ((size_t)(b * kC + c0 + cloc)) * NPCH + pc;
        pmax[off] = bestV;
        pidx[off] = bestI;
    }
}

// ---------------------------------------------------------------------------
// Kernel 2: reduce partial argmaxes over chunks; ReLU rule (max<=0 -> idx 0).
// ---------------------------------------------------------------------------
__global__ void reduce_argmax_kernel(const float* __restrict__ pmax,
                                     const int* __restrict__ pidx,
                                     int* __restrict__ idxout)
{
    const int i = blockIdx.x * blockDim.x + threadIdx.x;  // b*kC + c
    if (i >= kB * kC) return;
    float bv = -__builtin_inff();
    int   bi = 0;
    for (int pc = 0; pc < NPCH; ++pc) {
        const float v  = pmax[(size_t)i * NPCH + pc];
        const int   id = pidx[(size_t)i * NPCH + pc];
        if (v > bv || (v == bv && id < bi)) { bv = v; bi = id; }
    }
    idxout[i] = (bv > 0.0f) ? bi : 0;
}

// ---------------------------------------------------------------------------
// Kernel 3: zero the output (runs AFTER the GEMM has consumed the bf16
// scratch planes living in d_out).
// ---------------------------------------------------------------------------
__global__ void zero_out_kernel(float4* __restrict__ out, int n4)
{
    const int i = blockIdx.x * blockDim.x + threadIdx.x;
    if (i < n4) out[i] = make_float4(0.f, 0.f, 0.f, 0.f);
}

// ---------------------------------------------------------------------------
// Kernel 4: scatter-add kernel rows into winning positions (f32 atomics).
// ---------------------------------------------------------------------------
__global__ __launch_bounds__(256) void scatter_add_kernel(
    const float* __restrict__ kern, const int* __restrict__ idx,
    float* __restrict__ out)
{
    const int bc = blockIdx.x;            // b*kC + c
    const int b  = bc / kC;
    const int c  = bc % kC;
    const int p  = idx[bc];
    const int d  = threadIdx.x;
    atomicAdd(&out[((size_t)b * kP + p) * kD + d], kern[c * kD + d]);
}

extern "C" void kernel_launch(void* const* d_in, const int* in_sizes, int n_in,
                              void* d_out, int out_size, void* d_ws, size_t ws_size,
                              hipStream_t stream)
{
    const float* x    = (const float*)d_in[0];   // (16, 4096, 256) f32
    const float* kern = (const float*)d_in[1];   // (512, 256) f32
    float* out = (float*)d_out;                  // (16, 4096, 256) f32

    // bf16 hi/lo planes of x live in d_out until the GEMM has consumed them
    // (16M + 16M bf16 = 64 MB = exactly sizeof(out)); zero-fill comes after.
    __bf16* xh = (__bf16*)d_out;
    __bf16* xl = xh + (size_t)kB * kP * kD;

    // workspace: partial maxima / indices / final indices (~1.03 MB)
    float* pmax = (float*)d_ws;
    int*   pidx = (int*)((char*)d_ws + (size_t)kB * kC * NPCH * 4);
    int*   idxf = (int*)((char*)d_ws + (size_t)2 * kB * kC * NPCH * 4);

    const int n4 = kB * kP * kD / 4;
    split_x_kernel<<<(n4 + 255) / 256, 256, 0, stream>>>((const float4*)x, xh,
                                                         xl, n4);

    dim3 g1(kB * CB, NPCH);
    gemm_argmax_kernel<<<g1, 64, 0, stream>>>(xh, xl, kern, pmax, pidx);

    reduce_argmax_kernel<<<(kB * kC + 255) / 256, 256, 0, stream>>>(pmax, pidx,
                                                                    idxf);

    zero_out_kernel<<<(n4 + 255) / 256, 256, 0, stream>>>((float4*)out, n4);

    scatter_add_kernel<<<kB * kC, kD, 0, stream>>>(kern, idxf, out);
}
